// RSNN_12979391169197
// MI455X (gfx1250) — matrix operand order repaired and checked by reading, not compiled, not run on hardware
//
#include <hip/hip_runtime.h>

// ---------------------------------------------------------------------------
// RSNN forward scan for MI455X (gfx1250, wave32, WMMA).
//
// N_DELAY=5 synaptic delay => steps [t0, t0+4] depend only on spikes from
// earlier kernels, so each launch advances 5 timesteps:
//   i_t = [z(t-5) | x_t] @ [w_rec ; w_in]   (bf16 WMMA, f32 acc)
// with per-neuron v/b/r state held in registers across the 5 inner steps.
// Each weight fragment is loaded once and feeds 5 back-to-back WMMAs.
// Spike ring and x are stored pre-swizzled in the WMMA A-fragment layout so
// every GEMM load is 32 contiguous bytes per lane (coalesced).
// Forward z is exactly binary (surrogate term cancels) -> bf16 A is lossless.
// ---------------------------------------------------------------------------

typedef __attribute__((ext_vector_type(16))) __bf16 v16bf;
typedef __attribute__((ext_vector_type(8)))  float  v8f;

#define N_UNITS 2000
#define EXC     1700
#define BATCH   32
#define TSTEPS  1500
#define NIN     200
#define KREC    2016                 // 2000 padded to mult of 32
#define KIN     224                  // 200 padded to mult of 32
#define NCHUNK  70                   // (KREC+KIN)/32
#define ZCHUNK  63                   // KREC/32
#define XCHUNK  7                    // KIN/32
#define NTILES  125                  // 2000/16 column tiles
#define NDELAY  5
#define SPK     5                    // steps per kernel (== NDELAY)
#define NRING   16                   // ring slots (read/write windows disjoint mod 16)
#define RSLOT   (ZCHUNK * 2 * 512)   // 64512 bf16 per ring slot (frag layout)
#define XSLOT   (XCHUNK * 2 * 512)   // 7168 bf16 per timestep (frag layout)

#define DECAY_EXC 0.95122945f        // exp(-1/20)
#define DECAY_INH 0.90483743f        // exp(-1/10)
#define RHO_AD    0.99307961f        // exp(-1/144)
#define ONE_M_RHO 0.00692039f

// workspace byte offsets (all 256B aligned)
static constexpr size_t OFF_WSW  = 0;          // swizzled weights bf16: 70*125*512 el = 8,960,000 B
static constexpr size_t OFF_XBF  = 8960000;    // x frag bf16 [T][7][2][32][16]  = 21,504,000 B
static constexpr size_t OFF_RING = 30464000;   // z ring bf16 [16][63][2][32][16]=  2,064,384 B
static constexpr size_t OFF_V    = 32528384;   // v state f32 [125][32][16]      =    256,000 B
static constexpr size_t OFF_B    = 32784384;   // b state f32                    =    256,000 B
static constexpr size_t OFF_R    = 33040384;   // r state i32                    =    256,000 B

// A-fragment element coords for matrix entry (row m, K=k):
//   mt = m>>4, lane = (m&15) + 16*((k>>3)&1), j = (k&7) + 8*((k&31)>=16)
// fragment index: ((kc*2 + mt)*32 + lane)*16 + j,  kc = k>>5.

// --------------------------- prep kernels ----------------------------------

// Swizzle [w_rec ; w_in] (f32) into per-lane WMMA B-fragment layout, bf16.
// Fragment (kc, nt): lane l holds column n = nt*16 + (l&15),
// elements j=0..15 -> K = kc*32 + 8*(l>=16) + (j&7) + 16*(j>>3), stored contiguously (32B/lane).
__global__ __launch_bounds__(256) void prep_w(const float* __restrict__ w_rec,
                                              const float* __restrict__ w_in,
                                              __bf16* __restrict__ wsw) {
  int idx = blockIdx.x * 256 + threadIdx.x;
  if (idx >= NCHUNK * NTILES * 512) return;
  int j    = idx & 15;
  int lane = (idx >> 4) & 31;
  int frag = idx >> 9;
  int nt   = frag % NTILES;
  int kc   = frag / NTILES;
  int n = nt * 16 + (lane & 15);
  int k = kc * 32 + ((lane & 16) ? 8 : 0) + (j & 7) + ((j >> 3) << 4);
  float val = 0.0f;
  if (k < N_UNITS)            val = w_rec[(size_t)k * N_UNITS + n];
  else if (k >= KREC && k < KREC + NIN)
                              val = w_in[(size_t)(k - KREC) * N_UNITS + n];
  wsw[idx] = (__bf16)val;
}

// Convert input spikes to bf16 directly in A-fragment layout [T][7][2][32][16].
__global__ __launch_bounds__(256) void prep_x(const float* __restrict__ inp,
                                              __bf16* __restrict__ xbf) {
  int idx = blockIdx.x * 256 + threadIdx.x;
  if (idx >= TSTEPS * XSLOT) return;
  int j    = idx & 15;
  int lane = (idx >> 4) & 31;
  int q    = idx >> 9;
  int mtg  = q & 1;
  int q2   = q >> 1;
  int kcx  = q2 % XCHUNK;
  int t    = q2 / XCHUNK;
  int m = (lane & 15) + 16 * mtg;
  int k = kcx * 32 + ((lane & 16) ? 8 : 0) + (j & 7) + ((j >> 3) << 4);
  float val = (k < NIN) ? inp[((size_t)t * BATCH + m) * NIN + k] : 0.0f;
  xbf[idx] = (__bf16)val;
}

// Zero ring + v/b/r state (workspace is poisoned 0xAA; must re-init each call).
__global__ __launch_bounds__(256) void prep_state(float* __restrict__ vs,
                                                  float* __restrict__ bs,
                                                  int* __restrict__ rs,
                                                  unsigned* __restrict__ ring32) {
  int idx = blockIdx.x * 256 + threadIdx.x;
  if (idx < NRING * RSLOT / 2) ring32[idx] = 0u;         // 516096 u32
  if (idx < NTILES * 512) { vs[idx] = 0.0f; bs[idx] = 0.0f; rs[idx] = 0; }
}

// --------------------------- 5-step kernel ---------------------------------

__global__ __launch_bounds__(256) void rsnn_step5(const __bf16* __restrict__ wsw,
                                                  const __bf16* __restrict__ xbf,
                                                  __bf16* __restrict__ ring,
                                                  float* __restrict__ vstate,
                                                  float* __restrict__ bstate,
                                                  int* __restrict__ rstate,
                                                  const float* __restrict__ noise,
                                                  float* __restrict__ out,
                                                  int t0) {
  __shared__ float red[8 * 256];                 // per-wave partial tile (one mt each)
  const int tid  = threadIdx.x;
  const int lane = tid & 31;
  // wave-uniform values forced scalar -> SALU loop control, scalar branches
  const int wave = __builtin_amdgcn_readfirstlane(tid >> 5);
  const int w4   = wave & 3;                     // K-split index within mt group
  const int mt   = wave >> 2;                    // 0: batch rows 0..15, 1: rows 16..31
  const int nt   = blockIdx.x;

  // Scalar per-step A bases (mt offset folded in); lane part is lane*16.
  size_t sZ[SPK], sX[SPK];
#pragma unroll
  for (int s = 0; s < SPK; ++s) {
    sZ[s] = (size_t)((t0 + s - NDELAY) & (NRING - 1)) * RSLOT + (size_t)mt * 512;
    sX[s] = (size_t)(t0 + s) * XSLOT + (size_t)mt * 512;
  }
  const size_t laneOff = (size_t)lane * 16;
  const __bf16* bbase  = wsw + (size_t)nt * 512 + laneOff;   // + kc*64000 per chunk

  v8f acc[SPK] = {};

  // ---- GEMM phase 1: recurrent K (A = z(t-5), fragment-layout ring) -------
  for (int kc = w4; kc < ZCHUNK; kc += 4) {
    union { v16bf v; uint4 u[2]; } bf;
    const uint4* bp = (const uint4*)(bbase + (size_t)kc * (NTILES * 512));
    bf.u[0] = bp[0];
    bf.u[1] = bp[1];
    __builtin_prefetch(bbase + (size_t)(kc + 4) * (NTILES * 512), 0, 1);

    union { v16bf v; uint4 u[2]; } af[SPK];
#pragma unroll
    for (int s = 0; s < SPK; ++s) {
      const uint4* ap = (const uint4*)(ring + sZ[s] + (size_t)kc * 1024 + laneOff);
      af[s].u[0] = ap[0];
      af[s].u[1] = ap[1];
    }
#pragma unroll
    for (int s = 0; s < SPK; ++s)
      acc[s] = __builtin_amdgcn_wmma_f32_16x16x32_bf16(false, af[s].v, false, bf.v,
                                                       (short)0, acc[s], false, false);
  }

  // ---- GEMM phase 2: input K (A = x_t, fragment-layout) -------------------
  for (int kcx = (w4 + 1) & 3; kcx < XCHUNK; kcx += 4) {
    union { v16bf v; uint4 u[2]; } bf;
    const uint4* bp = (const uint4*)(bbase + (size_t)(ZCHUNK + kcx) * (NTILES * 512));
    bf.u[0] = bp[0];
    bf.u[1] = bp[1];

    union { v16bf v; uint4 u[2]; } af[SPK];
#pragma unroll
    for (int s = 0; s < SPK; ++s) {
      const uint4* ap = (const uint4*)(xbf + sX[s] + (size_t)kcx * 1024 + laneOff);
      af[s].u[0] = ap[0];
      af[s].u[1] = ap[1];
    }
#pragma unroll
    for (int s = 0; s < SPK; ++s)
      acc[s] = __builtin_amdgcn_wmma_f32_16x16x32_bf16(false, af[s].v, false, bf.v,
                                                       (short)0, acc[s], false, false);
  }

  // ---- state phase: reduce partials + 5 sequential neuron updates ---------
  // Thread owns 2 elements: o = tid and tid+256.
  // C layout: (VGPR j, lane l) of tile mt -> (M = j + 8*(l>=16) + 16*mt, N = l&15)
  int    mA[2], nA[2], rS[2];
  float  vS[2], bS[2];
  size_t sidxA[2], ringOff[2];
#pragma unroll
  for (int e = 0; e < 2; ++e) {
    int o   = tid + e * 256;
    int l   = o & 31;
    int j   = (o >> 5) & 7;
    int mtE = o >> 8;
    mA[e] = j + ((l & 16) ? 8 : 0) + mtE * 16;
    nA[e] = nt * 16 + (l & 15);
    sidxA[e] = (size_t)nt * 512 + mA[e] * 16 + (nA[e] & 15);
    vS[e] = vstate[sidxA[e]];
    bS[e] = bstate[sidxA[e]];
    rS[e] = rstate[sidxA[e]];
    // A-fragment coords of (row mA, K=nA) for the spike-ring write
    int kcW   = nA[e] >> 5;
    int koW   = nA[e] & 31;
    int laneW = (mA[e] & 15) + 16 * ((koW >> 3) & 1);
    int jW    = (koW & 7) + 8 * (koW >= 16);
    ringOff[e] = ((size_t)(kcW * 2 + (mA[e] >> 4)) * 32 + laneW) * 16 + jW;
  }

#pragma unroll
  for (int s = 0; s < SPK; ++s) {
    __syncthreads();                             // red free from previous step
#pragma unroll
    for (int j = 0; j < 8; ++j)
      red[wave * 256 + j * 32 + lane] = acc[s][j];
    __syncthreads();

    const int t = t0 + s;
#pragma unroll
    for (int e = 0; e < 2; ++e) {
      int o   = tid + e * 256;
      int mtE = o >> 8;
      int oo  = o & 255;
      float sum = red[(mtE * 4 + 0) * 256 + oo] + red[(mtE * 4 + 1) * 256 + oo]
                + red[(mtE * 4 + 2) * 256 + oo] + red[(mtE * 4 + 3) * 256 + oo];

      size_t gidx = (size_t)t * (BATCH * N_UNITS) + (size_t)mA[e] * N_UNITS + nA[e];
      float v = vS[e];
      float b = bS[e];
      int   r = rS[e];

      float dv = (nA[e] < EXC) ? DECAY_EXC : DECAY_INH;
      v = dv * v + sum + 0.01f * noise[gidx];    // sigma*(thr-vrest)=0.01, vrest=0
      float thr = 0.1f + 1.6f * b * ((nA[e] % 5 == 0) ? 1.0f : 0.0f);
      float z = ((v > thr) && (r == 0)) ? 1.0f : 0.0f;   // forward surrogate == 0
      v -= z * 0.1f;                             // soft reset
      b = RHO_AD * b + ONE_M_RHO * z;
      r = (z > 0.5f) ? 4 : (r > 0 ? r - 1 : 0);

      vS[e] = v; bS[e] = b; rS[e] = r;
      out[gidx] = z;
      ring[(size_t)(t & (NRING - 1)) * RSLOT + ringOff[e]] = (__bf16)z;
    }
  }

#pragma unroll
  for (int e = 0; e < 2; ++e) {
    vstate[sidxA[e]] = vS[e];
    bstate[sidxA[e]] = bS[e];
    rstate[sidxA[e]] = rS[e];
  }
}

// ---------------------------------------------------------------------------

extern "C" void kernel_launch(void* const* d_in, const int* in_sizes, int n_in,
                              void* d_out, int out_size, void* d_ws, size_t ws_size,
                              hipStream_t stream) {
  (void)in_sizes; (void)n_in; (void)out_size; (void)ws_size;

  const float* inp   = (const float*)d_in[0];   // [T,B,200]
  const float* w_in  = (const float*)d_in[1];   // [200,2000]
  const float* w_rec = (const float*)d_in[2];   // [2000,2000]
  const float* noise = (const float*)d_in[3];   // [T,B,2000]

  char*    ws   = (char*)d_ws;
  __bf16*  wsw  = (__bf16*)(ws + OFF_WSW);
  __bf16*  xbf  = (__bf16*)(ws + OFF_XBF);
  __bf16*  ring = (__bf16*)(ws + OFF_RING);
  float*   vs   = (float*)(ws + OFF_V);
  float*   bs   = (float*)(ws + OFF_B);
  int*     rs   = (int*)(ws + OFF_R);
  float*   out  = (float*)d_out;

  prep_w    <<<(NCHUNK * NTILES * 512 + 255) / 256, 256, 0, stream>>>(w_rec, w_in, wsw);
  prep_x    <<<(TSTEPS * XSLOT + 255) / 256, 256, 0, stream>>>(inp, xbf);
  prep_state<<<(NRING * RSLOT / 2 + 255) / 256, 256, 0, stream>>>(vs, bs, rs,
                                                                  (unsigned*)ring);

  for (int t0 = 0; t0 < TSTEPS; t0 += SPK)
    rsnn_step5<<<NTILES, 256, 0, stream>>>(wsw, xbf, ring, vs, bs, rs, noise, out, t0);
}
